// DCRNNModel_28243704938815
// MI455X (gfx1250) — matrix operand order repaired and checked by reading, not compiled, not run on hardware
//
#include <hip/hip_runtime.h>
#include <hip/hip_bf16.h>

// ---------------- problem constants ----------------
#define BB 64      // batch
#define TT 8       // encoder timesteps
#define NN 1000    // nodes
#define UU 64      // hidden units
#define FF 65      // concat feature size (1 + U)
#define FP 96      // F padded to 3 x 32 (WMMA K-steps)
#define NP 1024    // N padded to 32 x 32
#define CW (BB*FP) // 6144 feature columns
#define K3 3       // Chebyshev order

typedef __attribute__((ext_vector_type(16))) __bf16 v16bf;
typedef __attribute__((ext_vector_type(8)))  float  v8f;

union Frag { v16bf bf; unsigned short us[16]; unsigned int ui[8]; };

__device__ __forceinline__ unsigned short f2bf(float f) {
  unsigned int u = __float_as_uint(f);
  u += 0x7FFFu + ((u >> 16) & 1u);     // round-to-nearest-even
  return (unsigned short)(u >> 16);
}
__device__ __forceinline__ float bf2f(unsigned short s) {
  return __uint_as_float(((unsigned int)s) << 16);
}

// A fragment: 16x32 bf16 tile of row-major A, rows row0..row0+15, K k0..k0+31.
// ISA: lane l<16 -> row l, K = {kh*8+0..7, 16+kh*8+0..7}, kh = lane>>4.
// All K-pairs contiguous -> 2x16B packed loads per lane.
__device__ __forceinline__ void load_a(Frag& fr, const unsigned short* __restrict__ A,
                                       int lda, int row0, int k0, int lane) {
  int l = lane & 15, kh = (lane >> 4) & 1;
  const unsigned short* p = A + (size_t)(row0 + l) * lda + k0 + kh * 8;
#pragma unroll
  for (int j = 0; j < 4; ++j) fr.ui[j] = *(const unsigned int*)(p + 2 * j);
  p += 16;
#pragma unroll
  for (int j = 0; j < 4; ++j) fr.ui[4 + j] = *(const unsigned int*)(p + 2 * j);
}

// B fragment of B = M^T where M is row-major (ldm): B[k][col] = M[col][k].
// ISA B layout: lane l -> col0+l, K = {kh*8+0..7, 16+kh*8+0..7} (same as A).
// => lane reads 2x4 consecutive dwords of row (col0+l) of M. Fully packed.
__device__ __forceinline__ void load_bt(Frag& fr, const unsigned short* __restrict__ M,
                                        int ldm, int col0, int k0, int lane) {
  int l = lane & 15, kh = (lane >> 4) & 1;
  const unsigned short* p = M + (size_t)(col0 + l) * ldm + k0 + kh * 8;
#pragma unroll
  for (int j = 0; j < 8; ++j) fr.ui[j] = *(const unsigned int*)(p + 2 * j + (j >> 2) * 8);
}

__device__ __forceinline__ v8f wmma_bf16(const Frag& a, const Frag& b, v8f c) {
  return __builtin_amdgcn_wmma_f32_16x16x32_bf16(false, a.bf, false, b.bf,
                                                 (short)0, c, false, false);
}

// ---------------- prep kernels ----------------

// S (1000x1000 f32) -> zero-padded 1024x1024 bf16 (row-major; serves as S^T source too)
__global__ void conv_S(unsigned short* __restrict__ Sbf, const float* __restrict__ S) {
  int i = blockIdx.x * blockDim.x + threadIdx.x;
  if (i >= NP * NP) return;
  int m = i / NP, n = i % NP;
  float v = (m < NN && n < NN) ? S[m * NN + n] : 0.f;
  Sbf[i] = f2bf(v);
}

// de-interleave + transpose W (195 x O, rows f*3+k) -> WT (3, O, FP) bf16, f zero-padded
__global__ void deint_wT(const float* __restrict__ src, unsigned short* __restrict__ dst, int O) {
  int i = blockIdx.x * blockDim.x + threadIdx.x;
  if (i >= K3 * O * FP) return;
  int s = i / (O * FP);
  int r = i % (O * FP);
  int o = r / FP, f = r % FP;
  float v = (f < FF) ? src[(f * K3 + s) * O + o] : 0.f;
  dst[i] = f2bf(v);
}

__global__ void zero_f32(float* __restrict__ p, int n) {
  int i = blockIdx.x * blockDim.x + threadIdx.x;
  if (i < n) p[i] = 0.f;
}

// Build feature panel in both layouts:
//   XT[c][n] (feature-major, stride NP)  and  XN[n][c] (node-major, stride CW)
// value: f==0 -> x_t (or 0), f in 1..64 -> (R?)*h, else 0.
__global__ void build_x0(unsigned short* __restrict__ XT, unsigned short* __restrict__ XN,
                         const float* __restrict__ xin, int t,
                         const float* __restrict__ H, const float* __restrict__ R) {
  int i = blockIdx.x * blockDim.x + threadIdx.x;
  if (i >= CW * NP) return;
  int c = i / NP, n = i % NP;
  int b = c / FP, f = c % FP;
  float v = 0.f;
  if (n < NN) {
    if (f == 0) {
      if (xin) v = xin[(b * TT + t) * NN + n];
    } else if (f <= UU) {
      int idx = (b * NN + n) * UU + (f - 1);
      v = H[idx];
      if (R) v *= R[idx];
    }
  }
  unsigned short h = f2bf(v);
  XT[i] = h;               // coalesced
  XN[n * CW + c] = h;      // scattered, small vs GEMM traffic
}

// ---------------- diffusion GEMM (transposed): YT = XT @ S^T  (=(S@X)^T)
// mode 1: YT = 2*XT1@S^T - X0T. Writes both YT and YN layouts.
// One wave computes a 64x64 output block (4x4 WMMA tiles), K = 1024:
// per k-step 8 fragment loads (16x b128) feed 16 WMMAs -> 1 b128 per WMMA.
// __launch_bounds__(32,1): full VGPR budget, no accumulator spills.
__global__ void __launch_bounds__(32, 1)
diff_gemm(const unsigned short* __restrict__ S,
          const unsigned short* __restrict__ XT,
          unsigned short* __restrict__ YT,
          unsigned short* __restrict__ YN,
          const unsigned short* __restrict__ XsubT, int mode) {
  int lane = threadIdx.x & 31;
  int m0 = blockIdx.x * 64;   // node block (output cols of YT)
  int c0 = blockIdx.y * 64;   // feature-col block (output rows of YT)
  v8f acc[4][4];
#pragma unroll
  for (int ia = 0; ia < 4; ++ia)
#pragma unroll
    for (int ib = 0; ib < 4; ++ib)
#pragma unroll
      for (int q = 0; q < 8; ++q) acc[ia][ib][q] = 0.f;

  for (int k0 = 0; k0 < NP; k0 += 32) {
    Frag a[4];
#pragma unroll
    for (int ia = 0; ia < 4; ++ia) load_a(a[ia], XT, NP, c0 + 16 * ia, k0, lane);
#pragma unroll
    for (int ib = 0; ib < 4; ++ib) {
      Frag b;
      load_bt(b, S, NP, m0 + 16 * ib, k0, lane);   // B = S^T from row-major S
#pragma unroll
      for (int ia = 0; ia < 4; ++ia) acc[ia][ib] = wmma_bf16(a[ia], b, acc[ia][ib]);
    }
  }

  int l = lane & 15, mh = (lane >> 4) & 1;
#pragma unroll
  for (int ia = 0; ia < 4; ++ia) {
#pragma unroll
    for (int ib = 0; ib < 4; ++ib) {
      int mcol = m0 + ib * 16 + l;
#pragma unroll
      for (int j = 0; j < 8; ++j) {
        int crow = c0 + ia * 16 + mh * 8 + j;
        float v = acc[ia][ib][j];
        if (mode) v = 2.f * v - bf2f(XsubT[crow * NP + mcol]);
        unsigned short h = f2bf(v);
        YT[crow * NP + mcol] = h;
        YN[mcol * CW + crow] = h;
      }
    }
  }
}

// ---------------- r/u GEMM: pre = sum_s XsN[:,b] @ Ws + bias; sigmoid; split R,U
// wave computes 32 rows x 64 cols (2x4 tiles).
__global__ void __launch_bounds__(32, 1)
ru_gemm(const unsigned short* __restrict__ X0N,
        const unsigned short* __restrict__ X1N,
        const unsigned short* __restrict__ X2N,
        const unsigned short* __restrict__ WT,  // (3, 128, FP) bf16
        const float* __restrict__ bias,         // 128
        float* __restrict__ R, float* __restrict__ Uu) {
  const int O = 2 * UU;
  int lane = threadIdx.x & 31;
  int og0 = blockIdx.x * 64;          // 0 or 64
  int n0  = blockIdx.y * 32;
  int b   = blockIdx.z;
  v8f acc[2][4];
#pragma unroll
  for (int ir = 0; ir < 2; ++ir)
#pragma unroll
    for (int io = 0; io < 4; ++io)
#pragma unroll
      for (int q = 0; q < 8; ++q) acc[ir][io][q] = 0.f;

  const unsigned short* Xs[3] = {X0N, X1N, X2N};
#pragma unroll
  for (int s = 0; s < 3; ++s) {
    const unsigned short* A = Xs[s] + b * FP;
    const unsigned short* W = WT + s * O * FP;
#pragma unroll
    for (int k0 = 0; k0 < FP; k0 += 32) {
      Frag a0, a1;
      load_a(a0, A, CW, n0,      k0, lane);
      load_a(a1, A, CW, n0 + 16, k0, lane);
#pragma unroll
      for (int io = 0; io < 4; ++io) {
        Frag w;
        load_bt(w, W, FP, og0 + io * 16, k0, lane);
        acc[0][io] = wmma_bf16(a0, w, acc[0][io]);
        acc[1][io] = wmma_bf16(a1, w, acc[1][io]);
      }
    }
  }

  int l = lane & 15, mh = (lane >> 4) & 1;
#pragma unroll
  for (int ir = 0; ir < 2; ++ir) {
#pragma unroll
    for (int io = 0; io < 4; ++io) {
      int o = og0 + io * 16 + l;
      float bi = bias[o];
#pragma unroll
      for (int j = 0; j < 8; ++j) {
        int row = n0 + ir * 16 + mh * 8 + j;
        if (row < NN) {
          float sg = 1.f / (1.f + __expf(-(acc[ir][io][j] + bi)));
          int idx = (b * NN + row) * UU + (o & 63);
          if (o < UU) R[idx] = sg; else Uu[idx] = sg;
        }
      }
    }
  }
}

// ---------------- c GEMM + GRU update: c = tanh(sum_s Xs@Ws + b); h' = u*h + (1-u)*c
// wave computes 32 rows x 64 cols (2x4 tiles).
__global__ void __launch_bounds__(32, 1)
c_gemm(const unsigned short* __restrict__ X0N,
       const unsigned short* __restrict__ X1N,
       const unsigned short* __restrict__ X2N,
       const unsigned short* __restrict__ WT,  // (3, 64, FP) bf16
       const float* __restrict__ bias,         // 64
       const float* __restrict__ Uu,
       const float* __restrict__ Hin,
       float* __restrict__ Hout) {
  const int O = UU;
  int lane = threadIdx.x & 31;
  int n0 = blockIdx.y * 32;
  int b  = blockIdx.z;
  v8f acc[2][4];
#pragma unroll
  for (int ir = 0; ir < 2; ++ir)
#pragma unroll
    for (int io = 0; io < 4; ++io)
#pragma unroll
      for (int q = 0; q < 8; ++q) acc[ir][io][q] = 0.f;

  const unsigned short* Xs[3] = {X0N, X1N, X2N};
#pragma unroll
  for (int s = 0; s < 3; ++s) {
    const unsigned short* A = Xs[s] + b * FP;
    const unsigned short* W = WT + s * O * FP;
#pragma unroll
    for (int k0 = 0; k0 < FP; k0 += 32) {
      Frag a0, a1;
      load_a(a0, A, CW, n0,      k0, lane);
      load_a(a1, A, CW, n0 + 16, k0, lane);
#pragma unroll
      for (int io = 0; io < 4; ++io) {
        Frag w;
        load_bt(w, W, FP, io * 16, k0, lane);
        acc[0][io] = wmma_bf16(a0, w, acc[0][io]);
        acc[1][io] = wmma_bf16(a1, w, acc[1][io]);
      }
    }
  }

  int l = lane & 15, mh = (lane >> 4) & 1;
#pragma unroll
  for (int ir = 0; ir < 2; ++ir) {
#pragma unroll
    for (int io = 0; io < 4; ++io) {
      int o = io * 16 + l;
      float bi = bias[o];
#pragma unroll
      for (int j = 0; j < 8; ++j) {
        int row = n0 + ir * 16 + mh * 8 + j;
        if (row < NN) {
          float c = tanhf(acc[ir][io][j] + bi);
          int idx = (b * NN + row) * UU + o;
          float u = Uu[idx], hv = Hin[idx];
          Hout[idx] = u * hv + (1.f - u) * c;
        }
      }
    }
  }
}

// ---------------- projection: out[b,n] = h . W_proj + b_proj
__global__ void proj(const float* __restrict__ H, const float* __restrict__ Wp,
                     const float* __restrict__ bp, float* __restrict__ out) {
  int i = blockIdx.x * blockDim.x + threadIdx.x;
  if (i >= BB * NN) return;
  float a = bp[0];
#pragma unroll
  for (int u = 0; u < UU; ++u) a += H[i * UU + u] * Wp[u];
  out[i] = a;
}

// ---------------- host launch ----------------
extern "C" void kernel_launch(void* const* d_in, const int* in_sizes, int n_in,
                              void* d_out, int out_size, void* d_ws, size_t ws_size,
                              hipStream_t stream) {
  (void)in_sizes; (void)n_in; (void)out_size; (void)ws_size;
  const float* inputs   = (const float*)d_in[0];
  const float* support  = (const float*)d_in[1];
  const float* enc_W_ru = (const float*)d_in[2];
  const float* enc_b_ru = (const float*)d_in[3];
  const float* enc_W_c  = (const float*)d_in[4];
  const float* enc_b_c  = (const float*)d_in[5];
  const float* dec_W_ru = (const float*)d_in[6];
  const float* dec_b_ru = (const float*)d_in[7];
  const float* dec_W_c  = (const float*)d_in[8];
  const float* dec_b_c  = (const float*)d_in[9];
  const float* W_proj   = (const float*)d_in[10];
  const float* b_proj   = (const float*)d_in[11];
  float* out = (float*)d_out;

  char* ws = (char*)d_ws;
  size_t off = 0;
  auto alloc = [&](size_t bytes) -> void* {
    void* p = ws + off;
    off = (off + bytes + 255) & ~(size_t)255;
    return p;
  };
  unsigned short* Sbf   = (unsigned short*)alloc((size_t)NP * NP * 2);
  unsigned short* X0T   = (unsigned short*)alloc((size_t)NP * CW * 2);
  unsigned short* X1T   = (unsigned short*)alloc((size_t)NP * CW * 2);
  unsigned short* X2T   = (unsigned short*)alloc((size_t)NP * CW * 2);
  unsigned short* X0N   = (unsigned short*)alloc((size_t)NP * CW * 2);
  unsigned short* X1N   = (unsigned short*)alloc((size_t)NP * CW * 2);
  unsigned short* X2N   = (unsigned short*)alloc((size_t)NP * CW * 2);
  unsigned short* Wru_e = (unsigned short*)alloc((size_t)K3 * 128 * FP * 2);
  unsigned short* Wc_e  = (unsigned short*)alloc((size_t)K3 * 64 * FP * 2);
  unsigned short* Wru_d = (unsigned short*)alloc((size_t)K3 * 128 * FP * 2);
  unsigned short* Wc_d  = (unsigned short*)alloc((size_t)K3 * 64 * FP * 2);
  const size_t HN = (size_t)BB * NN * UU;
  float* R  = (float*)alloc(HN * 4);
  float* Uu = (float*)alloc(HN * 4);
  float* Ha = (float*)alloc(HN * 4);
  float* Hb = (float*)alloc(HN * 4);

  const int TB = 256;
  conv_S<<<(NP * NP + TB - 1) / TB, TB, 0, stream>>>(Sbf, support);
  deint_wT<<<(K3 * 128 * FP + TB - 1) / TB, TB, 0, stream>>>(enc_W_ru, Wru_e, 128);
  deint_wT<<<(K3 * 64 * FP + TB - 1) / TB, TB, 0, stream>>>(enc_W_c, Wc_e, 64);
  deint_wT<<<(K3 * 128 * FP + TB - 1) / TB, TB, 0, stream>>>(dec_W_ru, Wru_d, 128);
  deint_wT<<<(K3 * 64 * FP + TB - 1) / TB, TB, 0, stream>>>(dec_W_c, Wc_d, 64);
  zero_f32<<<((int)HN + TB - 1) / TB, TB, 0, stream>>>(Ha, (int)HN);

  const dim3 gPanel((CW * NP + TB - 1) / TB);
  const dim3 gDiff(NP / 64, CW / 64);      // 16 x 96 blocks, 1 wave each, 512 WMMA/wave
  const dim3 gRu(2, NP / 32, BB);          // 2 x 32 x 64
  const dim3 gC(1, NP / 32, BB);           // 1 x 32 x 64

  auto cell = [&](const float* x, int t, const float* hin, float* hout,
                  const unsigned short* Wru, const float* bru,
                  const unsigned short* Wc, const float* bc) {
    build_x0<<<gPanel, TB, 0, stream>>>(X0T, X0N, x, t, hin, nullptr);
    diff_gemm<<<gDiff, 32, 0, stream>>>(Sbf, X0T, X1T, X1N, nullptr, 0);
    diff_gemm<<<gDiff, 32, 0, stream>>>(Sbf, X1T, X2T, X2N, X0T, 1);
    ru_gemm<<<gRu, 32, 0, stream>>>(X0N, X1N, X2N, Wru, bru, R, Uu);
    build_x0<<<gPanel, TB, 0, stream>>>(X0T, X0N, x, t, hin, R);  // x0 with r*h
    diff_gemm<<<gDiff, 32, 0, stream>>>(Sbf, X0T, X1T, X1N, nullptr, 0);
    diff_gemm<<<gDiff, 32, 0, stream>>>(Sbf, X1T, X2T, X2N, X0T, 1);
    c_gemm<<<gC, 32, 0, stream>>>(X0N, X1N, X2N, Wc, bc, Uu, hin, hout);
  };

  for (int t = 0; t < TT; ++t) {
    float* hin  = (t & 1) ? Hb : Ha;
    float* hout = (t & 1) ? Ha : Hb;
    cell(inputs, t, hin, hout, Wru_e, enc_b_ru, Wc_e, enc_b_c);
  }
  // h_enc in Ha after t=7; decoder with GO = zeros
  cell(nullptr, 0, Ha, Hb, Wru_d, dec_b_ru, Wc_d, dec_b_c);

  proj<<<(BB * NN + TB - 1) / TB, TB, 0, stream>>>(Hb, W_proj, b_proj, out);
}
